// LHMambaVisionMixer_11682311045747
// MI455X (gfx1250) — compile-verified, matrix operand order checked by reading, not run on hardware
//
#include <hip/hip_runtime.h>
#include <hip/hip_bf16.h>
#include <math.h>
#include <stdint.h>

// ---------------------------------------------------------------------------
// Types for CDNA5 WMMA (wave32, 16x16x32 bf16 -> f32 accum)
// ---------------------------------------------------------------------------
typedef __bf16 v16bf __attribute__((ext_vector_type(16)));
typedef float  v8f   __attribute__((ext_vector_type(8)));
typedef unsigned v4u __attribute__((ext_vector_type(4)));
typedef unsigned v8u __attribute__((ext_vector_type(8)));

#define BDIM   4
#define LSEQ   1024
#define DMODEL 768
#define DCI    256
#define NSTATE 16
#define RRANK  48
#define NHEAD  8
#define HDIM   32
#define BLROWS (BDIM * LSEQ)   // 4096

__device__ __forceinline__ int koff_fn(int j) { return j + (j & 8); }  // 0..7 -> 0..7, 8..15 -> 16..23

__device__ __forceinline__ float softplus_f(float v) {
    return (v > 20.0f) ? v : log1pf(__expf(v));
}
__device__ __forceinline__ float silu_f(float v) {
    return v / (1.0f + __expf(-v));
}

// Async global->LDS copy of 16 bytes (CDNA5, tracked by ASYNCcnt)
__device__ __forceinline__ void async_g2l_b128(unsigned lds_addr, unsigned long long gaddr) {
    asm volatile("global_load_async_to_lds_b128 %0, %1, off"
                 :: "v"(lds_addr), "v"(gaddr) : "memory");
}

// ---------------------------------------------------------------------------
// Tensor Data Mover: 2D tile load Global -> LDS (tracked by TENSORcnt).
// D# group0 (4 SGPRs): count=1 | lds_addr | global_addr[56:0] | type=2
// D# group1 (8 SGPRs): data_size | tensor_dim0/1 | tile_dim0/1 | dim0_stride
// Tile is written contiguously to LDS (row-major [tile_d1][tile_d0]).
// OOB reads (x >= tensor_d0) return zero -> hardware K-tail masking.
// ---------------------------------------------------------------------------
__device__ __forceinline__ void tdm_load_2d(unsigned lds_addr, const void* gptr,
                                            unsigned data_size_log,      // 0:1B 1:2B 2:4B
                                            unsigned tile_d0, unsigned tile_d1,
                                            unsigned tens_d0, unsigned tens_d1,
                                            unsigned long long stride0)  // elements
{
    unsigned long long ga = (unsigned long long)(uintptr_t)gptr;
    v4u g0;
    g0[0] = 1u;                                                  // count=1, user desc
    g0[1] = lds_addr;
    g0[2] = (unsigned)(ga & 0xFFFFFFFFu);
    g0[3] = (unsigned)((ga >> 32) & 0x01FFFFFFu) | (2u << 30);   // type=2 ("image")
    v8u g1;
    g1[0] = (data_size_log & 3u) << 16;
    g1[1] = (tens_d0 & 0xFFFFu) << 16;
    g1[2] = ((tens_d0 >> 16) & 0xFFFFu) | ((tens_d1 & 0xFFFFu) << 16);
    g1[3] = ((tens_d1 >> 16) & 0xFFFFu) | ((tile_d0 & 0xFFFFu) << 16);
    g1[4] = (tile_d1 & 0xFFFFu);                                 // tile_dim2 = 0
    g1[5] = (unsigned)(stride0 & 0xFFFFFFFFu);
    g1[6] = (unsigned)((stride0 >> 32) & 0xFFFFu);               // dim1_stride unused (2D)
    g1[7] = 0u;
    asm volatile("tensor_load_to_lds %0, %1" :: "s"(g0), "s"(g1) : "memory");
}

// ---------------------------------------------------------------------------
// fp32 -> bf16 weight conversion
// ---------------------------------------------------------------------------
__global__ void f32_to_bf16_kernel(const float* __restrict__ src, __bf16* __restrict__ dst, int n) {
    int t = blockIdx.x * blockDim.x + threadIdx.x;
    if (t < n) dst[t] = (__bf16)src[t];
}

// ---------------------------------------------------------------------------
// Block-tiled WMMA GEMM, double-buffered LDS staging:
//   A tile 64x32 fp32 via GLOBAL_LOAD_ASYNC_TO_LDS_B128 (ASYNCcnt)
//   W tile 64x32 bf16 via TENSOR_LOAD_TO_LDS            (TENSORcnt, wave 0)
// 128 threads = 4 waves; block computes a 64x64 tile; K-step 32.
// MODE: 0 none, 1 +bias, 2 softplus(x + 2*bias)
// ---------------------------------------------------------------------------
template <int MODE>
__global__ void gemm_bf16_wmma_tiled(const float* __restrict__ A, int lda,
                                     const __bf16* __restrict__ W,
                                     float* __restrict__ C, int ldc, int colOffC,
                                     const float* __restrict__ bias,
                                     int M, int N, int K) {
    __shared__ float  Alds[2][64][32];
    __shared__ __bf16 Blds[2][64][32];

    const int t     = threadIdx.x;
    const int lane  = t & 31;
    const int wv    = t >> 5;          // wave id in block = m sub-tile
    const int half  = lane >> 4;
    const int ln    = lane & 15;
    const int kbase = half * 8;

    const int nBlocks = N >> 6;
    const int mBase = (blockIdx.x / nBlocks) * 64;
    const int nBase = (blockIdx.x % nBlocks) * 64;
    const int nSteps = (K + 31) >> 5;

    auto stage = [&](int ks, int bi) {
        const int k0 = ks * 32;
        // A tile: 512 x b128 -> 4 per thread (async, ASYNCcnt)
#pragma unroll
        for (int i = 0; i < 4; ++i) {
            int chunk = t * 4 + i;        // 0..511
            int row   = chunk >> 3;
            int col   = (chunk & 7) * 4;
            unsigned lds = (unsigned)(uintptr_t)&Alds[bi][row][col];
            unsigned long long g =
                (unsigned long long)(uintptr_t)(A + (size_t)(mBase + row) * lda + k0 + col);
            async_g2l_b128(lds, g);
        }
        // W tile: one TDM descriptor per stage (wave 0 only; EXEC ignored by TDM)
        if (wv == 0) {
            tdm_load_2d((unsigned)(uintptr_t)&Blds[bi][0][0],
                        W + (size_t)nBase * K + k0,
                        /*data_size=*/1u, /*tile_d0=*/32u, /*tile_d1=*/64u,
                        /*tens_d0=*/(unsigned)(K - k0), /*tens_d1=*/(unsigned)(N - nBase),
                        /*stride0=*/(unsigned long long)(unsigned)K);
        }
    };

    v8f acc[4];
#pragma unroll
    for (int tt = 0; tt < 4; ++tt)
#pragma unroll
        for (int r = 0; r < 8; ++r) acc[tt][r] = 0.0f;

    stage(0, 0);

    for (int ks = 0; ks < nSteps; ++ks) {
        const int bi = ks & 1;
        if (ks + 1 < nSteps) {
            stage(ks + 1, bi ^ 1);
            // allow only the prefetched stage's 4 async loads / 1 TDM op in flight
            asm volatile("s_wait_asynccnt 0x4" ::: "memory");
            if (wv == 0) __builtin_amdgcn_s_wait_tensorcnt(1);
        } else {
            asm volatile("s_wait_asynccnt 0x0" ::: "memory");
            if (wv == 0) __builtin_amdgcn_s_wait_tensorcnt(0);
        }
        __syncthreads();

        const int k0 = ks * 32;
        v16bf afrag;
#pragma unroll
        for (int j = 0; j < 16; ++j) {
            int kk = kbase + koff_fn(j);
            float v = (k0 + kk < K) ? Alds[bi][wv * 16 + ln][kk] : 0.0f;
            afrag[j] = (__bf16)v;
        }
#pragma unroll
        for (int tt = 0; tt < 4; ++tt) {
            v16bf bfrag;
#pragma unroll
            for (int j = 0; j < 16; ++j) {
                int kk = kbase + koff_fn(j);
                bfrag[j] = Blds[bi][tt * 16 + ln][kk];   // TDM zero-fills K tail
            }
            acc[tt] = __builtin_amdgcn_wmma_f32_16x16x32_bf16(
                false, afrag, false, bfrag, (short)0, acc[tt], false, false);
        }
        __syncthreads();   // buffer bi free for reuse at stage ks+2
    }

#pragma unroll
    for (int tt = 0; tt < 4; ++tt) {
        int n = nBase + tt * 16 + ln;
        float bv = (MODE != 0) ? bias[n] : 0.0f;
#pragma unroll
        for (int r = 0; r < 8; ++r) {
            int m = mBase + wv * 16 + r + half * 8;
            float v = acc[tt][r];
            if (MODE == 1) v += bv;
            if (MODE == 2) v = softplus_f(v + 2.0f * bv);
            C[(size_t)m * ldc + colOffC + n] = v;
        }
    }
}

// ---------------------------------------------------------------------------
// Legacy per-wave WMMA GEMM (used for the N=80 x_proj). One wave = 16x80 tile.
// ---------------------------------------------------------------------------
template <int NT, int MODE>
__global__ void gemm_bf16_wmma(const float* __restrict__ A, int lda, int colOffA,
                               const __bf16* __restrict__ W,
                               float* __restrict__ C, int ldc, int colOffC,
                               const float* __restrict__ bias,
                               int M, int N, int K) {
    const int lane = threadIdx.x & 31;
    const int wave = blockIdx.x * (blockDim.x >> 5) + (threadIdx.x >> 5);
    const int mTiles  = M >> 4;
    const int nGroups = N / (16 * NT);
    if (wave >= mTiles * nGroups) return;
    const int mt = wave % mTiles;
    const int ng = wave / mTiles;

    const int half  = lane >> 4;
    const int ln    = lane & 15;
    const int kbase = half * 8;

    const float* aPtr = A + (size_t)(mt * 16 + ln) * lda + colOffA;

    v8f acc[NT];
#pragma unroll
    for (int t = 0; t < NT; ++t)
#pragma unroll
        for (int r = 0; r < 8; ++r) acc[t][r] = 0.0f;

    for (int k0 = 0; k0 < K; k0 += 32) {
        if (k0 + 64 < K) __builtin_prefetch(aPtr + k0 + 64, 0, 1);
        v16bf afrag;
#pragma unroll
        for (int j = 0; j < 16; ++j) {
            int k = k0 + kbase + koff_fn(j);
            float v = (k < K) ? aPtr[k] : 0.0f;
            afrag[j] = (__bf16)v;
        }
#pragma unroll
        for (int t = 0; t < NT; ++t) {
            int n = (ng * NT + t) * 16 + ln;
            const __bf16* wPtr = W + (size_t)n * K;
            v16bf bfrag;
#pragma unroll
            for (int j = 0; j < 16; ++j) {
                int k = k0 + kbase + koff_fn(j);
                bfrag[j] = (k < K) ? wPtr[k] : (__bf16)0.0f;
            }
            acc[t] = __builtin_amdgcn_wmma_f32_16x16x32_bf16(
                false, afrag, false, bfrag, (short)0, acc[t], false, false);
        }
    }

#pragma unroll
    for (int t = 0; t < NT; ++t) {
        int n = (ng * NT + t) * 16 + ln;
        float bv = (MODE != 0) ? bias[n] : 0.0f;
#pragma unroll
        for (int r = 0; r < 8; ++r) {
            int m = mt * 16 + r + half * 8;
            float v = acc[t][r];
            if (MODE == 1) v += bv;
            if (MODE == 2) v = softplus_f(v + 2.0f * bv);
            C[(size_t)m * ldc + colOffC + n] = v;
        }
    }
}

// ---------------------------------------------------------------------------
// Depthwise conv (K=4, pad left 1 / right 2) + SiLU
// ---------------------------------------------------------------------------
__global__ void dwconv_silu_kernel(const float* __restrict__ in, int ldin, int coIn,
                                   const float* __restrict__ w,
                                   float* __restrict__ out, int ldout, int coOut) {
    int t = blockIdx.x * blockDim.x + threadIdx.x;
    if (t >= BLROWS * DCI) return;
    int d   = t & (DCI - 1);
    int row = t >> 8;
    int l   = row & (LSEQ - 1);
    const float* wp = w + d * 4;
    float acc = 0.0f;
#pragma unroll
    for (int j = 0; j < 4; ++j) {
        int lp = l - 1 + j;
        if (lp >= 0 && lp < LSEQ)
            acc += in[(size_t)(row - l + lp) * ldin + coIn + d] * wp[j];
    }
    out[(size_t)row * ldout + coOut + d] = silu_f(acc);
}

// ---------------------------------------------------------------------------
// Selective scan: one wave handles 2 channels; lanes within a 16-lane half
// hold the 16 states. Writes x_out into cat-buffer columns [0,256).
// ---------------------------------------------------------------------------
__global__ void scan_kernel(const float* __restrict__ delta, const float* __restrict__ xconv,
                            const float* __restrict__ xdbl, const float* __restrict__ A_log,
                            const float* __restrict__ Dp, float* __restrict__ cat) {
    int wave = blockIdx.x * (blockDim.x >> 5) + (threadIdx.x >> 5);
    if (wave >= BDIM * (DCI / 2)) return;
    int b     = wave >> 7;
    int dpair = wave & 127;
    int lane  = threadIdx.x & 31;
    int half  = lane >> 4;
    int n     = lane & 15;
    int d     = dpair * 2 + half;

    float An = -__expf(A_log[d * NSTATE + n]);
    float Dd = Dp[d];
    float h  = 0.0f;
    size_t base = (size_t)b * LSEQ;
    for (int l = 0; l < LSEQ; ++l) {
        size_t row = base + l;
        float dt = delta[row * DCI + d];
        float xv = xconv[row * DCI + d];
        float Bn = xdbl[row * 80 + RRANK + n];
        float Cn = xdbl[row * 80 + RRANK + NSTATE + n];
        h = __expf(dt * An) * h + dt * xv * Bn;
        float y = h * Cn;
        y += __shfl_xor(y, 1, 32);
        y += __shfl_xor(y, 2, 32);
        y += __shfl_xor(y, 4, 32);
        y += __shfl_xor(y, 8, 32);
        if (n == 0) cat[row * (3 * DCI) + d] = y + xv * Dd;
    }
}

// ---------------------------------------------------------------------------
// Attention: one wave per (b, h, 16-query tile). HD=32 == one bf16 WMMA K step.
// V tiles double-buffered into LDS via async loads for the P*V phase.
// ---------------------------------------------------------------------------
__global__ void attention_kernel(const float* __restrict__ qkvb, float* __restrict__ obuf) {
    __shared__ __bf16 S[16][LSEQ + 8];
    __shared__ float  Vlds[2][32][32];
    __shared__ float  rmaxLds[16];

    int blk = blockIdx.x;
    int qt  = blk & 63;
    int h   = (blk >> 6) & (NHEAD - 1);
    int b   = blk >> 9;

    const int lane  = threadIdx.x & 31;
    const int half  = lane >> 4;
    const int ln    = lane & 15;
    const int kbase = half * 8;
    const float scale = 0.17677669529663689f;  // 1/sqrt(32)

    v16bf qf;
    {
        int g = b * LSEQ + qt * 16 + ln;
        const float* qrow = qkvb + (size_t)g * (3 * DCI) + h * HDIM;
#pragma unroll
        for (int j = 0; j < 16; ++j)
            qf[j] = (__bf16)(qrow[kbase + koff_fn(j)] * scale);
    }

    float rm[8];
#pragma unroll
    for (int r = 0; r < 8; ++r) rm[r] = -3.0e38f;

    for (int kt = 0; kt < 64; ++kt) {
        v16bf kf;
        int key = kt * 16 + ln;
        const float* krow = qkvb + (size_t)(b * LSEQ + key) * (3 * DCI) + DCI + h * HDIM;
#pragma unroll
        for (int j = 0; j < 16; ++j)
            kf[j] = (__bf16)krow[kbase + koff_fn(j)];

        v8f c;
#pragma unroll
        for (int r = 0; r < 8; ++r) c[r] = 0.0f;
        c = __builtin_amdgcn_wmma_f32_16x16x32_bf16(false, qf, false, kf, (short)0, c, false, false);

#pragma unroll
        for (int r = 0; r < 8; ++r) {
            float v = c[r];
            rm[r] = fmaxf(rm[r], v);
            S[r + half * 8][kt * 16 + ln] = (__bf16)v;
        }
    }

#pragma unroll
    for (int r = 0; r < 8; ++r) {
        rm[r] = fmaxf(rm[r], __shfl_xor(rm[r], 1, 32));
        rm[r] = fmaxf(rm[r], __shfl_xor(rm[r], 2, 32));
        rm[r] = fmaxf(rm[r], __shfl_xor(rm[r], 4, 32));
        rm[r] = fmaxf(rm[r], __shfl_xor(rm[r], 8, 32));
    }
    if (ln == 0) {
#pragma unroll
        for (int r = 0; r < 8; ++r) rmaxLds[half * 8 + r] = rm[r];
    }
    __syncthreads();

    float rmax[16];
#pragma unroll
    for (int m = 0; m < 16; ++m) rmax[m] = rmaxLds[m];

    float rsum[16];
#pragma unroll
    for (int m = 0; m < 16; ++m) rsum[m] = 0.0f;

    for (int c0 = lane; c0 < LSEQ; c0 += 32) {
#pragma unroll
        for (int m = 0; m < 16; ++m) {
            float s = (float)S[m][c0];
            float p = __expf(s - rmax[m]);
            S[m][c0] = (__bf16)p;
            rsum[m] += p;
        }
    }
#pragma unroll
    for (int m = 0; m < 16; ++m) {
        rsum[m] += __shfl_xor(rsum[m], 1, 32);
        rsum[m] += __shfl_xor(rsum[m], 2, 32);
        rsum[m] += __shfl_xor(rsum[m], 4, 32);
        rsum[m] += __shfl_xor(rsum[m], 8, 32);
        rsum[m] += __shfl_xor(rsum[m], 16, 32);
    }
    float rinv[16];
#pragma unroll
    for (int m = 0; m < 16; ++m) rinv[m] = 1.0f / rsum[m];
    __syncthreads();

    // ---- O = P V with async-staged, double-buffered V tiles ----
    auto stageV = [&](int kt2, int bi) {
#pragma unroll
        for (int i = 0; i < 8; ++i) {
            int c   = i * 32 + lane;      // 0..255 chunks of 16B
            int row = c >> 3;             // key within tile
            int col = (c & 7) * 4;        // hd offset
            unsigned lds = (unsigned)(uintptr_t)&Vlds[bi][row][col];
            unsigned long long g = (unsigned long long)(uintptr_t)(
                qkvb + (size_t)(b * LSEQ + kt2 * 32 + row) * (3 * DCI) + 2 * DCI + h * HDIM + col);
            async_g2l_b128(lds, g);
        }
    };

    v8f o0, o1;
#pragma unroll
    for (int r = 0; r < 8; ++r) { o0[r] = 0.0f; o1[r] = 0.0f; }

    stageV(0, 0);
    for (int kt2 = 0; kt2 < 32; ++kt2) {
        const int bi = kt2 & 1;
        if (kt2 + 1 < 32) {
            stageV(kt2 + 1, bi ^ 1);
            asm volatile("s_wait_asynccnt 0x8" ::: "memory");  // current tile complete
        } else {
            asm volatile("s_wait_asynccnt 0x0" ::: "memory");
        }

        v16bf af;
#pragma unroll
        for (int j = 0; j < 16; ++j)
            af[j] = S[ln][kt2 * 32 + kbase + koff_fn(j)];

        v16bf v0f, v1f;
#pragma unroll
        for (int j = 0; j < 16; ++j) {
            int kk = kbase + koff_fn(j);
            v0f[j] = (__bf16)Vlds[bi][kk][ln];
            v1f[j] = (__bf16)Vlds[bi][kk][16 + ln];
        }
        o0 = __builtin_amdgcn_wmma_f32_16x16x32_bf16(false, af, false, v0f, (short)0, o0, false, false);
        o1 = __builtin_amdgcn_wmma_f32_16x16x32_bf16(false, af, false, v1f, (short)0, o1, false, false);
    }

#pragma unroll
    for (int r = 0; r < 8; ++r) {
        int m = r + half * 8;
        float inv = rinv[m];
        size_t row = (size_t)(b * LSEQ + qt * 16 + m);
        obuf[row * DCI + h * HDIM + ln]      = o0[r] * inv;
        obuf[row * DCI + h * HDIM + 16 + ln] = o1[r] * inv;
    }
}

// ---------------------------------------------------------------------------
// Host-side orchestration
// ---------------------------------------------------------------------------
static inline size_t align256(size_t x) { return (x + 255) & ~(size_t)255; }

extern "C" void kernel_launch(void* const* d_in, const int* in_sizes, int n_in,
                              void* d_out, int out_size, void* d_ws, size_t ws_size,
                              hipStream_t stream) {
    (void)in_sizes; (void)n_in; (void)out_size; (void)ws_size;

    const float* hidden   = (const float*)d_in[0];
    const float* inW      = (const float*)d_in[1];
    const float* convWx   = (const float*)d_in[2];
    const float* convWz   = (const float*)d_in[3];
    const float* convWy   = (const float*)d_in[4];
    const float* xpW      = (const float*)d_in[5];
    const float* dtW      = (const float*)d_in[6];
    const float* dtB      = (const float*)d_in[7];
    const float* A_log    = (const float*)d_in[8];
    const float* Dp       = (const float*)d_in[9];
    const float* outW     = (const float*)d_in[10];
    const float* qkvW     = (const float*)d_in[11];
    const float* atW      = (const float*)d_in[12];
    const float* atB      = (const float*)d_in[13];
    float* out = (float*)d_out;

    char* ws = (char*)d_ws;
    size_t off = 0;
    auto take = [&](size_t bytes) { size_t o = off; off = align256(off + bytes); return o; };

    size_t o_wIn   = take((size_t)768 * 768 * 2);
    size_t o_wXp   = take((size_t)80  * 256 * 2);
    size_t o_wDt   = take((size_t)256 * 48  * 2);
    size_t o_wQkv  = take((size_t)768 * 256 * 2);
    size_t o_wAt   = take((size_t)256 * 256 * 2);
    size_t o_wOut  = take((size_t)768 * 768 * 2);
    size_t o_xzy   = take((size_t)BLROWS * 768 * 4);  // later reused as qkv buffer
    size_t o_xconv = take((size_t)BLROWS * 256 * 4);  // later reused as attn output
    size_t o_yconv = take((size_t)BLROWS * 256 * 4);
    size_t o_xdbl  = take((size_t)BLROWS * 80  * 4);
    size_t o_delta = take((size_t)BLROWS * 256 * 4);
    size_t o_cat   = take((size_t)BLROWS * 768 * 4);

    __bf16* wIn  = (__bf16*)(ws + o_wIn);
    __bf16* wXp  = (__bf16*)(ws + o_wXp);
    __bf16* wDt  = (__bf16*)(ws + o_wDt);
    __bf16* wQkv = (__bf16*)(ws + o_wQkv);
    __bf16* wAt  = (__bf16*)(ws + o_wAt);
    __bf16* wOut = (__bf16*)(ws + o_wOut);
    float* xzy   = (float*)(ws + o_xzy);
    float* xconv = (float*)(ws + o_xconv);
    float* yconv = (float*)(ws + o_yconv);
    float* xdbl  = (float*)(ws + o_xdbl);
    float* delta = (float*)(ws + o_delta);
    float* catb  = (float*)(ws + o_cat);
    float* qkvb  = xzy;    // alias (xzy dead after convs)
    float* obuf  = xconv;  // alias (xconv dead after scan)

    auto cvt = [&](const float* s, __bf16* d, int n) {
        f32_to_bf16_kernel<<<(n + 255) / 256, 256, 0, stream>>>(s, d, n);
    };
    cvt(inW,  wIn,  768 * 768);
    cvt(xpW,  wXp,  80 * 256);
    cvt(dtW,  wDt,  256 * 48);
    cvt(qkvW, wQkv, 768 * 256);
    cvt(atW,  wAt,  256 * 256);
    cvt(outW, wOut, 768 * 768);

    auto tiles = [](int M, int N) { return dim3((M / 64) * (N / 64)); };

    // 1) xzy = hidden @ in_proj_w^T                (4096 x 768, K=768)
    gemm_bf16_wmma_tiled<0><<<tiles(BLROWS, 768), 128, 0, stream>>>(
        hidden, 768, wIn, xzy, 768, 0, nullptr, BLROWS, 768, 768);

    // 2) depthwise convs + SiLU (x -> xconv, z -> cat[:,256:512], y -> yconv)
    dim3 cgrid((BLROWS * DCI + 255) / 256);
    dwconv_silu_kernel<<<cgrid, 256, 0, stream>>>(xzy, 768, 0,   convWx, xconv, 256, 0);
    dwconv_silu_kernel<<<cgrid, 256, 0, stream>>>(xzy, 768, 256, convWz, catb,  768, 256);
    dwconv_silu_kernel<<<cgrid, 256, 0, stream>>>(xzy, 768, 512, convWy, yconv, 256, 0);

    // 3) x_dbl = xconv @ x_proj_w^T               (4096 x 80, K=256) - N not /64
    {
        int waves = (BLROWS / 16);
        gemm_bf16_wmma<5, 0><<<dim3((waves + 3) / 4), 128, 0, stream>>>(
            xconv, 256, 0, wXp, xdbl, 80, 0, nullptr, BLROWS, 80, 256);
    }

    // 4) delta = softplus(dt @ dt_proj_w^T + 2*b) (4096 x 256, K=48)
    gemm_bf16_wmma_tiled<2><<<tiles(BLROWS, 256), 128, 0, stream>>>(
        xdbl, 80, wDt, delta, 256, 0, dtB, BLROWS, 256, 48);

    // 5) selective scan -> cat[:,0:256]
    scan_kernel<<<64, 256, 0, stream>>>(delta, xconv, xdbl, A_log, Dp, catb);

    // 6) qkv = yconv @ qkv_w^T                    (4096 x 768, K=256)
    gemm_bf16_wmma_tiled<0><<<tiles(BLROWS, 768), 128, 0, stream>>>(
        yconv, 256, wQkv, qkvb, 768, 0, nullptr, BLROWS, 768, 256);

    // 7) attention -> obuf (4096 x 256)
    attention_kernel<<<BDIM * NHEAD * (LSEQ / 16), 32, 0, stream>>>(qkvb, obuf);

    // 8) y_att = obuf @ attn_proj_w^T + b -> cat[:,512:768]
    gemm_bf16_wmma_tiled<1><<<tiles(BLROWS, 256), 128, 0, stream>>>(
        obuf, 256, wAt, catb, 768, 512, atB, BLROWS, 256, 256);

    // 9) out = cat @ out_proj_w^T                 (4096 x 768, K=768)
    gemm_bf16_wmma_tiled<0><<<tiles(BLROWS, 768), 128, 0, stream>>>(
        catb, 768, wOut, out, 768, 0, nullptr, BLROWS, 768, 768);
}
